// MambaEEGModel_49770081026635
// MI455X (gfx1250) — matrix operand order repaired and checked by reading, not compiled, not run on hardware
//
#include <hip/hip_runtime.h>
#include <math.h>

#define D_MODEL 128
#define D_INNER 256
#define D_STATE 16
#define D_CONV  4
#define DT_RANK 8
#define NB 16
#define NL 512
#define NROWS (NB * NL)   /* 8192 */
#define W_LEN 2048
#define N_LAYERS 6

typedef __attribute__((ext_vector_type(16))) __bf16 v16bf;
typedef __attribute__((ext_vector_type(8)))  float  v8f;

static __device__ inline v8f vzero8() {
  v8f z = {0.f, 0.f, 0.f, 0.f, 0.f, 0.f, 0.f, 0.f};
  return z;
}

// ---- A-fragment loader (f32 row-major MxK source -> bf16 fragment) ---------
// ISA layout: lane m = lane%16; lane-half h adds +8 to K; VGPR i (i<4): K=2i..,
// i>=4: K=16+2(i-4).. ; packed pair (low=even K, high=odd K). Pairs are
// contiguous in memory -> float2 loads (k offsets even, ld even => 8B aligned).
static __device__ inline v16bf load_a_bf16(const float* __restrict__ A, int ld,
                                           int row0, int k0) {
  int lane = threadIdx.x & 31;
  int m = lane & 15;
  int h = lane >> 4;
  const float2* base =
      (const float2*)(A + (size_t)(row0 + m) * (size_t)ld + k0);
  v16bf a;
#pragma unroll
  for (int i = 0; i < 8; ++i) {
    int kb = ((i < 4) ? 0 : 16) + ((i & 3) * 2) + h * 8;  // even
    float2 v = base[kb >> 1];
    a[2 * i]     = (__bf16)v.x;
    a[2 * i + 1] = (__bf16)v.y;
  }
  return a;
}

// ---- Pre-packed B fragments ------------------------------------------------
// B-fragment ISA layout for 32x16 bf16: lane n = lane%16; element j of the
// fragment is K = half*8 + j (j<8), K = 16 + half*8 + (j-8) (j>=8), matching
// the A fragment. We pre-swizzle weights so each lane's 16 bf16 (32 bytes)
// are contiguous: pack[((ks*nn + nt)*32 + lane)*16 + j].
__global__ __launch_bounds__(256) void prepack_b_kernel(
    const float* __restrict__ W, int N, int nk, int nn,
    __bf16* __restrict__ out) {
  int idx = blockIdx.x * blockDim.x + threadIdx.x;
  int total = nk * nn * 32;
  if (idx >= total) return;
  int lane = idx & 31;
  int tile = idx >> 5;  // ks*nn + nt
  int nt = tile % nn;
  int ks = tile / nn;
  int n = nt * 16 + (lane & 15);
  int kbase = ks * 32 + (lane >> 4) * 8;
  __bf16* dst = out + (size_t)idx * 16;
#pragma unroll
  for (int j = 0; j < 16; ++j)
    dst[j] = (__bf16)W[(size_t)(kbase + j + ((j >= 8) ? 8 : 0)) * N + n];
}

static __device__ inline v16bf load_b_packed(const __bf16* __restrict__ P,
                                             int nn, int ks, int nt) {
  int lane = threadIdx.x & 31;
  const v16bf* p = (const v16bf*)P;
  return p[(size_t)(ks * nn + nt) * 32 + lane];
}

static __device__ inline v8f wmma_bf16(v16bf a, v16bf b, v8f c) {
  return __builtin_amdgcn_wmma_f32_16x16x32_bf16(false, a, false, b, (short)0, c,
                                                 false, false);
}

// ---------------------------------------------------------------------------
// Kernel 1: tok = LN(windows @ tok_w + tok_b); h = tok + temporal + spatial
// One wave per 16-row strip; 8 N-tiles (full 128 cols) so LN fuses in-block.
// ---------------------------------------------------------------------------
__global__ __launch_bounds__(32) void tok_embed_kernel(
    const float* __restrict__ windows,      // (8192, 2048)
    const __bf16* __restrict__ tok_w_pack,  // packed (2048,128), nn=8
    const float* __restrict__ tok_b, const float* __restrict__ tok_g,
    const float* __restrict__ tok_beta, const float* __restrict__ temp_w,
    const float* __restrict__ temp_b, const float* __restrict__ proj,  // (3,128)
    const float* __restrict__ coords,       // (16,3)
    const int* __restrict__ seq_lengths, float* __restrict__ hout)  // (8192,128)
{
  __shared__ float tile[16][D_MODEL];
  int lane = threadIdx.x;
  int row0 = blockIdx.x * 16;
  v8f acc[8];
#pragma unroll
  for (int t = 0; t < 8; ++t) acc[t] = vzero8();

  for (int ks = 0; ks < W_LEN / 32; ++ks) {
    v16bf a = load_a_bf16(windows, W_LEN, row0, ks * 32);
#pragma unroll
    for (int nt = 0; nt < 8; ++nt) {
      v16bf b = load_b_packed(tok_w_pack, 8, ks, nt);
      acc[nt] = wmma_bf16(a, b, acc[nt]);
    }
  }
  int n = lane & 15, hh = lane >> 4;
#pragma unroll
  for (int nt = 0; nt < 8; ++nt)
#pragma unroll
    for (int r = 0; r < 8; ++r)
      tile[r + 8 * hh][nt * 16 + n] = acc[nt][r] + tok_b[nt * 16 + n];
  __syncthreads();

  if (lane < 16) {
    int m = lane;
    int grow = row0 + m;
    int bb = grow >> 9;   // / 512
    int ll = grow & 511;
    float mean = 0.f;
    for (int j = 0; j < D_MODEL; ++j) mean += tile[m][j];
    mean *= (1.0f / D_MODEL);
    float var = 0.f;
    for (int j = 0; j < D_MODEL; ++j) {
      float d = tile[m][j] - mean;
      var += d * d;
    }
    var *= (1.0f / D_MODEL);
    float inv = rsqrtf(var + 1e-5f);
    int sl = seq_lengths[bb];
    float T = fmaxf((float)sl, 1.0f);
    float npos = (ll < sl) ? ((float)ll / T) : 0.0f;
    float c0 = coords[bb * 3 + 0], c1 = coords[bb * 3 + 1], c2 = coords[bb * 3 + 2];
    float nrm = sqrtf(c0 * c0 + c1 * c1 + c2 * c2);
    float s = 1.0f / fmaxf(nrm, 1e-12f);
    c0 *= s; c1 *= s; c2 *= s;
    for (int j = 0; j < D_MODEL; ++j) {
      float t = (tile[m][j] - mean) * inv * tok_g[j] + tok_beta[j];
      float temporal = npos * temp_w[j] + temp_b[j];
      float spatial = c0 * proj[j] + c1 * proj[D_MODEL + j] + c2 * proj[2 * D_MODEL + j];
      hout[(size_t)grow * D_MODEL + j] = t + temporal + spatial;
    }
  }
}

// ---------------------------------------------------------------------------
// in_proj: xz(8192,512) = h(8192,128) @ W(128,512). Wave does 16 rows x 64 cols.
// ---------------------------------------------------------------------------
__global__ __launch_bounds__(32) void inproj_kernel(
    const float* __restrict__ hin, const __bf16* __restrict__ Wpack,  // nn=32
    float* __restrict__ xz) {
  int row0 = blockIdx.x * 16;
  int nt0 = blockIdx.y * 4;
  v8f acc[4];
#pragma unroll
  for (int t = 0; t < 4; ++t) acc[t] = vzero8();
  for (int ks = 0; ks < D_MODEL / 32; ++ks) {
    v16bf a = load_a_bf16(hin, D_MODEL, row0, ks * 32);
#pragma unroll
    for (int nt = 0; nt < 4; ++nt) {
      v16bf b = load_b_packed(Wpack, 32, ks, nt0 + nt);
      acc[nt] = wmma_bf16(a, b, acc[nt]);
    }
  }
  int lane = threadIdx.x;
  int n = lane & 15, hh = lane >> 4;
#pragma unroll
  for (int nt = 0; nt < 4; ++nt)
#pragma unroll
    for (int r = 0; r < 8; ++r)
      xz[(size_t)(row0 + r + 8 * hh) * (2 * D_INNER) + (nt0 + nt) * 16 + n] =
          acc[nt][r];
}

// ---------------------------------------------------------------------------
// Depthwise causal conv(k=4) + bias + SiLU on xc = xz[..., :256].
// ---------------------------------------------------------------------------
__global__ __launch_bounds__(256) void conv_silu_kernel(
    const float* __restrict__ xz, const float* __restrict__ conv_w,
    const float* __restrict__ conv_b, float* __restrict__ xcp) {
  int idx = blockIdx.x * blockDim.x + threadIdx.x;  // over 8192*256
  if (idx >= NROWS * D_INNER) return;
  int d = idx & (D_INNER - 1);
  int row = idx >> 8;   // b*512 + l
  int l = row & (NL - 1);
  int b = row >> 9;
  float acc = conv_b[d];
#pragma unroll
  for (int j = 0; j < D_CONV; ++j) {
    int lsrc = l - (D_CONV - 1) + j;
    if (lsrc >= 0)
      acc += conv_w[d * D_CONV + j] *
             xz[((size_t)(b * NL + lsrc)) * (2 * D_INNER) + d];
  }
  float sig = 1.0f / (1.0f + expf(-acc));
  xcp[idx] = acc * sig;
}

// ---------------------------------------------------------------------------
// x_proj (256->40) + dt_proj (8->256) + softplus; split out B,C. Block per row.
// ---------------------------------------------------------------------------
__global__ __launch_bounds__(256) void xproj_dt_kernel(
    const float* __restrict__ xcp, const float* __restrict__ x_proj_w,  // (256,40)
    const float* __restrict__ dt_proj_w,  // (8,256)
    const float* __restrict__ dt_proj_b, float* __restrict__ dt_out,
    float* __restrict__ Bm_out, float* __restrict__ Cm_out) {
  __shared__ float xrow[D_INNER];
  __shared__ float dbc[DT_RANK + 2 * D_STATE];
  int row = blockIdx.x;
  int t = threadIdx.x;
  xrow[t] = xcp[(size_t)row * D_INNER + t];
  __syncthreads();
  if (t < DT_RANK + 2 * D_STATE) {
    float s = 0.f;
    for (int i = 0; i < D_INNER; ++i)
      s += xrow[i] * x_proj_w[i * (DT_RANK + 2 * D_STATE) + t];
    dbc[t] = s;
  }
  __syncthreads();
  float s = dt_proj_b[t];
#pragma unroll
  for (int r = 0; r < DT_RANK; ++r) s += dbc[r] * dt_proj_w[r * D_INNER + t];
  float dtv = (s > 20.f) ? s : log1pf(expf(s));   // softplus
  dt_out[(size_t)row * D_INNER + t] = dtv;
  if (t < D_STATE)
    Bm_out[(size_t)row * D_STATE + t] = dbc[DT_RANK + t];
  else if (t < 2 * D_STATE)
    Cm_out[(size_t)row * D_STATE + (t - D_STATE)] =
        dbc[DT_RANK + D_STATE + (t - D_STATE)];
}

// ---------------------------------------------------------------------------
// Selective scan: block b, thread d; 16-float state in registers, 512 steps.
// Fuses + xc*D and * silu(z).
// ---------------------------------------------------------------------------
__global__ __launch_bounds__(256) void scan_kernel(
    const float* __restrict__ dt, const float* __restrict__ Bm,
    const float* __restrict__ Cm, const float* __restrict__ xcp,
    const float* __restrict__ xz, const float* __restrict__ A_log,  // (256,16)
    const float* __restrict__ Dp, float* __restrict__ y_out) {
  int b = blockIdx.x;
  int d = threadIdx.x;
  float A[D_STATE];
#pragma unroll
  for (int s2 = 0; s2 < D_STATE; ++s2) A[s2] = -expf(A_log[d * D_STATE + s2]);
  float Dd = Dp[d];
  float hst[D_STATE];
#pragma unroll
  for (int s2 = 0; s2 < D_STATE; ++s2) hst[s2] = 0.f;
  __shared__ float sB[D_STATE], sC[D_STATE];
  for (int l = 0; l < NL; ++l) {
    size_t row = (size_t)b * NL + l;
    if (d < D_STATE) sB[d] = Bm[row * D_STATE + d];
    else if (d < 2 * D_STATE) sC[d - D_STATE] = Cm[row * D_STATE + (d - D_STATE)];
    __syncthreads();
    float dtv = dt[row * D_INNER + d];
    float xv = xcp[row * D_INNER + d];
    float zv = xz[row * (2 * D_INNER) + D_INNER + d];
    float y = 0.f;
#pragma unroll
    for (int s2 = 0; s2 < D_STATE; ++s2) {
      float dA = expf(dtv * A[s2]);
      hst[s2] = dA * hst[s2] + dtv * sB[s2] * xv;
      y += hst[s2] * sC[s2];
    }
    y += xv * Dd;
    float sil = zv / (1.0f + expf(-zv));
    y_out[row * D_INNER + d] = y * sil;
    __syncthreads();
  }
}

// ---------------------------------------------------------------------------
// out_proj + residual + LayerNorm, in-place on h. Wave per 16-row strip.
// ---------------------------------------------------------------------------
__global__ __launch_bounds__(32) void outproj_ln_kernel(
    const float* __restrict__ y, const __bf16* __restrict__ Wpack,  // nn=8
    const float* __restrict__ ln_g, const float* __restrict__ ln_b,
    float* __restrict__ h) {
  __shared__ float tile[16][D_MODEL];
  int lane = threadIdx.x;
  int row0 = blockIdx.x * 16;
  v8f acc[8];
#pragma unroll
  for (int t = 0; t < 8; ++t) acc[t] = vzero8();
  for (int ks = 0; ks < D_INNER / 32; ++ks) {
    v16bf a = load_a_bf16(y, D_INNER, row0, ks * 32);
#pragma unroll
    for (int nt = 0; nt < 8; ++nt) {
      v16bf b = load_b_packed(Wpack, 8, ks, nt);
      acc[nt] = wmma_bf16(a, b, acc[nt]);
    }
  }
  int n = lane & 15, hh = lane >> 4;
#pragma unroll
  for (int nt = 0; nt < 8; ++nt)
#pragma unroll
    for (int r = 0; r < 8; ++r) {
      int m = r + 8 * hh;
      tile[m][nt * 16 + n] =
          acc[nt][r] + h[(size_t)(row0 + m) * D_MODEL + nt * 16 + n];
    }
  __syncthreads();
  if (lane < 16) {
    int m = lane;
    float mean = 0.f;
    for (int j = 0; j < D_MODEL; ++j) mean += tile[m][j];
    mean *= (1.0f / D_MODEL);
    float var = 0.f;
    for (int j = 0; j < D_MODEL; ++j) {
      float d = tile[m][j] - mean;
      var += d * d;
    }
    var *= (1.0f / D_MODEL);
    float inv = rsqrtf(var + 1e-5f);
    for (int j = 0; j < D_MODEL; ++j)
      h[(size_t)(row0 + m) * D_MODEL + j] =
          (tile[m][j] - mean) * inv * ln_g[j] + ln_b[j];
  }
}

// ---------------------------------------------------------------------------
// Final LayerNorm -> d_out.
// ---------------------------------------------------------------------------
__global__ __launch_bounds__(128) void final_ln_kernel(
    const float* __restrict__ h, const float* __restrict__ g,
    const float* __restrict__ b, float* __restrict__ out) {
  __shared__ float red[D_MODEL];
  int row = blockIdx.x;
  int t = threadIdx.x;
  float v = h[(size_t)row * D_MODEL + t];
  red[t] = v;
  __syncthreads();
  for (int off = 64; off > 0; off >>= 1) {
    if (t < off) red[t] += red[t + off];
    __syncthreads();
  }
  float mean = red[0] * (1.0f / D_MODEL);
  __syncthreads();
  float d = v - mean;
  red[t] = d * d;
  __syncthreads();
  for (int off = 64; off > 0; off >>= 1) {
    if (t < off) red[t] += red[t + off];
    __syncthreads();
  }
  float var = red[0] * (1.0f / D_MODEL);
  float inv = rsqrtf(var + 1e-5f);
  out[(size_t)row * D_MODEL + t] = d * inv * g[t] + b[t];
}

// ---------------------------------------------------------------------------
// Launch: inputs are windows, coords, seq_lengths, then params as jax-tree
// leaves. Layer i leaves (sorted keys) start at 3 + i*11:
//   0 A_log, 1 D, 2 conv_b, 3 conv_w, 4 dt_proj_b, 5 dt_proj_w, 6 in_proj_w,
//   7 ln_b, 8 ln_g, 9 out_proj_w, 10 x_proj_w.
// Then 69 out_b, 70 out_g, 71 proj, 72 temp_b, 73 temp_w, 74 tok_b,
// 75 tok_beta, 76 tok_g, 77 tok_w.
// ---------------------------------------------------------------------------
extern "C" void kernel_launch(void* const* d_in, const int* in_sizes, int n_in,
                              void* d_out, int out_size, void* d_ws,
                              size_t ws_size, hipStream_t stream) {
  (void)in_sizes; (void)n_in; (void)out_size; (void)ws_size;
  const float* windows = (const float*)d_in[0];
  const float* coords  = (const float*)d_in[1];
  const int*   seqlen  = (const int*)d_in[2];
  auto LP = [&](int layer, int leaf) -> const float* {
    return (const float*)d_in[3 + layer * 11 + leaf];
  };
  const float* out_b    = (const float*)d_in[69];
  const float* out_g    = (const float*)d_in[70];
  const float* proj     = (const float*)d_in[71];
  const float* temp_b   = (const float*)d_in[72];
  const float* temp_w   = (const float*)d_in[73];
  const float* tok_b    = (const float*)d_in[74];
  const float* tok_beta = (const float*)d_in[75];
  const float* tok_g    = (const float*)d_in[76];
  const float* tok_w    = (const float*)d_in[77];

  // Workspace carve-up (256B aligned chunks).
  char* wsb = (char*)d_ws;
  size_t off = 0;
  auto alloc = [&](size_t bytes) -> void* {
    void* p = wsb + off;
    off += (bytes + 255) & ~(size_t)255;
    return p;
  };
  float* h    = (float*)alloc((size_t)NROWS * D_MODEL * 4);
  float* xz   = (float*)alloc((size_t)NROWS * 2 * D_INNER * 4);
  float* xcp  = (float*)alloc((size_t)NROWS * D_INNER * 4);
  float* dt   = (float*)alloc((size_t)NROWS * D_INNER * 4);
  float* Bm   = (float*)alloc((size_t)NROWS * D_STATE * 4);
  float* Cm   = (float*)alloc((size_t)NROWS * D_STATE * 4);
  float* y    = (float*)alloc((size_t)NROWS * D_INNER * 4);
  __bf16* ptok = (__bf16*)alloc((size_t)W_LEN * D_MODEL * 2);
  __bf16* pin[N_LAYERS];
  __bf16* pout[N_LAYERS];
  for (int i = 0; i < N_LAYERS; ++i) {
    pin[i]  = (__bf16*)alloc((size_t)D_MODEL * 2 * D_INNER * 2);
    pout[i] = (__bf16*)alloc((size_t)D_INNER * D_MODEL * 2);
  }

  // Pre-pack all weights into WMMA B-fragment layout (bf16).
  {
    int nthr = (W_LEN / 32) * (D_MODEL / 16) * 32;  // 16384
    prepack_b_kernel<<<(nthr + 255) / 256, 256, 0, stream>>>(
        tok_w, D_MODEL, W_LEN / 32, D_MODEL / 16, ptok);
  }
  for (int i = 0; i < N_LAYERS; ++i) {
    int nthr_in = (D_MODEL / 32) * ((2 * D_INNER) / 16) * 32;  // 4096
    prepack_b_kernel<<<(nthr_in + 255) / 256, 256, 0, stream>>>(
        LP(i, 6), 2 * D_INNER, D_MODEL / 32, (2 * D_INNER) / 16, pin[i]);
    int nthr_out = (D_INNER / 32) * (D_MODEL / 16) * 32;  // 2048
    prepack_b_kernel<<<(nthr_out + 255) / 256, 256, 0, stream>>>(
        LP(i, 9), D_MODEL, D_INNER / 32, D_MODEL / 16, pout[i]);
  }

  tok_embed_kernel<<<NROWS / 16, 32, 0, stream>>>(
      windows, ptok, tok_b, tok_g, tok_beta, temp_w, temp_b, proj, coords,
      seqlen, h);

  for (int i = 0; i < N_LAYERS; ++i) {
    inproj_kernel<<<dim3(NROWS / 16, 8), 32, 0, stream>>>(h, pin[i], xz);
    conv_silu_kernel<<<(NROWS * D_INNER) / 256, 256, 0, stream>>>(
        xz, LP(i, 3), LP(i, 2), xcp);
    xproj_dt_kernel<<<NROWS, 256, 0, stream>>>(xcp, LP(i, 10), LP(i, 5),
                                               LP(i, 4), dt, Bm, Cm);
    scan_kernel<<<NB, 256, 0, stream>>>(dt, Bm, Cm, xcp, xz, LP(i, 0), LP(i, 1),
                                        y);
    outproj_ln_kernel<<<NROWS / 16, 32, 0, stream>>>(y, pout[i], LP(i, 8),
                                                     LP(i, 7), h);
  }

  final_ln_kernel<<<NROWS, 128, 0, stream>>>(h, out_g, out_b, (float*)d_out);
}
